// AttentionRNNEncoder_29832842838367
// MI455X (gfx1250) — compile-verified, matrix operand order checked by reading, not run on hardware
//
#include <hip/hip_runtime.h>

// ---------------------------------------------------------------------------
// AttentionRNNEncoder for MI455X (gfx1250, wave32, WMMA + async-to-LDS)
//   P1: pack x -> f16 (32768x64, zero padded K)
//   P2: pack W0^T, W1^T -> f16 (256x64, zero padded K)
//   K1: XW0 = x @ W0 + b0      (WMMA f16->f32, async global->LDS staging)
//   K2: LSTM0 recurrence (h@U0 via WMMA, U0^T in LDS)    -> h_seq (f16)
//   K3: HW1 = h_seq @ W1 + b1  (WMMA)
//   K4: LSTM1 recurrence (h@U1 via WMMA)                 -> latent (f16)
//   K5: beta normalization
//   K6: attention: fp = x*kernel+bias; G = fp fp^T (WMMA 3x3 tiles);
//       Cholesky solve G a = fp@y; ohat = a^T fp; scale by beta  -> ohat buf
//   K7: z[b,u] = sum_t ohat[b,t,u]     (fixed-order, deterministic)
// ---------------------------------------------------------------------------

#define BSZ 256
#define TT  128
#define DFF 40
#define HH  64
#define G4  256   // 4*H
#define NBT (BSZ * TT)   // 32768 rows

typedef __attribute__((ext_vector_type(16))) _Float16 v16h;
typedef __attribute__((ext_vector_type(8)))  _Float16 v8h;
typedef __attribute__((ext_vector_type(8)))  float    v8f;

// ---- fast math (v_rcp_f32 / v_exp_f32 instead of IEEE div / libm) ---------
__device__ __forceinline__ float fast_rcp(float x) { return __builtin_amdgcn_rcpf(x); }
__device__ __forceinline__ float sigm(float v) {
  return fast_rcp(1.f + __builtin_amdgcn_exp2f(-1.44269504f * v));
}
__device__ __forceinline__ float tanh_fast(float v) {
  return __builtin_fmaf(2.f, sigm(2.f * v), -1.f);
}

// ---- async global->LDS copy of one 16B chunk (CDNA5, ASYNCcnt) ------------
__device__ __forceinline__ void async_copy16(const _Float16* gsrc, _Float16* ldst) {
  unsigned loff = (unsigned)(uintptr_t)(void*)ldst;   // LDS byte offset
  asm volatile("global_load_async_to_lds_b128 %0, %1, off"
               :: "v"(loff), "v"(gsrc) : "memory");
}
__device__ __forceinline__ void async_wait0() {
  asm volatile("s_wait_asynccnt 0x0" ::: "memory");
}

// ---------------------------------------------------------------------------
// Fragment builder for 16x16x32 f16 WMMA (CDNA5 ISA 7.12.2, wave32).
// Source is a K-contiguous LDS buffer with leading dim `ld` (multiple of 8):
//   A-fragment:        src = A   (M-major),  row0 = M tile base, col0 = K base
//   B-fragment (B^T):  src = B^T (N-major),  row0 = N tile base, col0 = K base
// Per-lane data = two 16-byte contiguous LDS loads (ds_load_b128).
// ---------------------------------------------------------------------------
__device__ __forceinline__ v16h frag_rowmajor(const _Float16* src, int ld,
                                              int row0, int col0) {
  const int lane = threadIdx.x & 31;
  const int r  = row0 + (lane & 15);
  const int kb = (lane & 16) ? 8 : 0;
  const _Float16* p = src + r * ld + col0 + kb;
  v8h lo = *(const v8h*)p;          // K = col0+kb .. +7
  v8h hi = *(const v8h*)(p + 16);   // K = col0+16+kb .. +7
  return __builtin_shufflevector(lo, hi, 0, 1, 2, 3, 4, 5, 6, 7,
                                         8, 9, 10, 11, 12, 13, 14, 15);
}

// ---- P1: pack rows f32(r x kmax) -> f16(r x 64), zero padded --------------
__global__ __launch_bounds__(256) void pack_rows_kernel(
    const float* __restrict__ src, _Float16* __restrict__ dst, int kmax) {
  const int idx = blockIdx.x * 256 + threadIdx.x;   // 0 .. NBT*64-1
  const int r = idx >> 6, k = idx & 63;
  dst[idx] = (_Float16)((k < kmax) ? src[r * kmax + k] : 0.f);
}

// ---- P2: pack W(kmax x 256) -> W^T f16 (256 x 64), zero padded ------------
__global__ __launch_bounds__(256) void packT_kernel(
    const float* __restrict__ W, _Float16* __restrict__ WT, int kmax) {
  const int idx = blockIdx.x * 256 + threadIdx.x;   // 0 .. 16383
  const int k = idx >> 8, n = idx & 255;
  WT[n * 64 + k] = (_Float16)((k < kmax) ? W[k * G4 + n] : 0.f);
}

// ---- K1/K3: C(32768x256) = AH(32768x64 f16) * WTg^T + bias[n] -------------
// Block: 256 threads (8 waves). WT (32 KB) async-staged once; 8 M-strips per
// block, A strip (2 KB) async-staged per iteration; wave does 2 N-tiles.
__global__ __launch_bounds__(256) void gemm_bias_kernel(
    const _Float16* __restrict__ AH, const _Float16* __restrict__ WTg,
    const float* __restrict__ bias, float* __restrict__ C) {
  __shared__ __align__(16) _Float16 WT[G4 * 64];     // 32 KB, [n][k]
  __shared__ __align__(16) _Float16 Astrip[16 * 64]; // 2 KB,  [m][k]
  const int tid  = threadIdx.x;
  const int lane = tid & 31;
  const int wave = tid >> 5;                     // 0..7
  // Stage W^T via async copies: 2048 x 16B chunks, 8 per thread.
#pragma unroll
  for (int c = 0; c < 8; ++c) {
    const int chunk = tid + c * 256;
    async_copy16(WTg + chunk * 8, &WT[chunk * 8]);
  }
  const int nl = lane & 15;
  const int mb = (lane & 16) ? 8 : 0;
  for (int it = 0; it < 8; ++it) {
    const int strip = blockIdx.x * 8 + it;       // 0..2047
    const int row0 = strip * 16;
    if (tid < 128) async_copy16(AH + row0 * 64 + tid * 8, &Astrip[tid * 8]);
    async_wait0();
    __syncthreads();
#pragma unroll
    for (int half = 0; half < 2; ++half) {       // wave covers 2 N-tiles
      const int n0 = (wave + half * 8) * 16;
      v8f acc = {};
#pragma unroll
      for (int ks = 0; ks < 64; ks += 32) {
        v16h a = frag_rowmajor(Astrip, 64, 0, ks);
        v16h b = frag_rowmajor(WT, 64, n0, ks);
        acc = __builtin_amdgcn_wmma_f32_16x16x32_f16(false, a, false, b,
                                                     (short)0, acc, false, false);
      }
      const int n = n0 + nl;
      const float bn = bias[n];
#pragma unroll
      for (int r = 0; r < 8; ++r)
        C[(row0 + r + mb) * G4 + n] = acc[r] + bn;
    }
    __syncthreads();   // Astrip consumed before next iteration overwrites
  }
}

// ---- K2/K4: LSTM recurrence. ZW holds x@W+b; U is HxG4; hseq out f16 ------
__global__ __launch_bounds__(512) void lstm_kernel(
    const float* __restrict__ ZW, const float* __restrict__ U,
    _Float16* __restrict__ hseq) {
  __shared__ __align__(16) _Float16 UT[G4 * HH];   // 32 KB, U^T: [n][k]
  __shared__ __align__(16) _Float16 hH[16 * HH];   // 2 KB,  [m][k]
  __shared__ float cS[16 * HH];                    // 4 KB
  __shared__ float zS[16 * G4];                    // 16 KB
  const int tid  = threadIdx.x;
  const int lane = tid & 31;
  const int wave = tid >> 5;          // 0..15 -> N tile
  const int bbase = blockIdx.x * 16;  // 16 batch rows per block
  for (int idx = tid; idx < G4 * HH; idx += 512) {
    const int k = idx >> 8, n = idx & 255;         // coalesced read of U[k][n]
    UT[n * HH + k] = (_Float16)U[k * G4 + n];
  }
  for (int i = tid; i < 16 * HH; i += 512) { hH[i] = (_Float16)0.f; cS[i] = 0.f; }
  __syncthreads();
  const int n0 = wave * 16;
  const int n  = n0 + (lane & 15);
  const int mb = (lane & 16) ? 8 : 0;
  for (int t = 0; t < TT; ++t) {
    v8f acc = {};
#pragma unroll
    for (int ks = 0; ks < 64; ks += 32) {
      v16h a = frag_rowmajor(hH, HH, 0, ks);
      v16h b = frag_rowmajor(UT, HH, n0, ks);
      acc = __builtin_amdgcn_wmma_f32_16x16x32_f16(false, a, false, b,
                                                   (short)0, acc, false, false);
    }
#pragma unroll
    for (int r = 0; r < 8; ++r) {
      const int m = r + mb;
      zS[m * G4 + n] = acc[r] + ZW[((bbase + m) * TT + t) * G4 + n];
    }
    __syncthreads();
    for (int idx = tid; idx < 16 * HH; idx += 512) {
      const int m = idx >> 6, j = idx & 63;
      const float ig = sigm(zS[m * G4 + j]);
      const float fg = sigm(zS[m * G4 + 64 + j]);
      const float gg = tanh_fast(zS[m * G4 + 128 + j]);
      const float og = sigm(zS[m * G4 + 192 + j]);
      const float c  = fg * cS[idx] + ig * gg;
      cS[idx] = c;
      const float h = og * tanh_fast(c);
      const _Float16 hh = (_Float16)h;
      hH[idx] = hh;
      hseq[((bbase + m) * TT + t) * HH + j] = hh;
    }
    __syncthreads();
  }
}

// ---- K5: beta = |bw|+1e-8 normalized over T --------------------------------
__global__ __launch_bounds__(128) void beta_kernel(const float* __restrict__ bw,
                                                   float* __restrict__ betaN) {
  __shared__ float s[TT];
  const int t = threadIdx.x;
  const float a = fabsf(bw[t]) + 1e-8f;
  s[t] = a;
  __syncthreads();
  for (int off = 64; off > 0; off >>= 1) {
    if (t < off) s[t] += s[t + off];
    __syncthreads();
  }
  betaN[t] = a * fast_rcp(s[0]);
}

// ---- K6: attention: per (b,t) build fp, G=fp fp^T (WMMA), Cholesky solve --
__global__ __launch_bounds__(128) void attn_kernel(
    const float* __restrict__ x, const _Float16* __restrict__ latent,
    const float* __restrict__ kern, const float* __restrict__ pbias,
    const float* __restrict__ betaN, float* __restrict__ ohat) {
  __shared__ __align__(16) _Float16 fpH[4][48 * 64]; // 24 KB (rows 40..47 = 0)
  __shared__ float Gm[4][40 * 41];                   // 25.6 KB
  __shared__ float yS[4][64];
  __shared__ float rhs[4][48];
  const int tid  = threadIdx.x;
  const int lane = tid & 31;
  const int w    = tid >> 5;             // wave -> instance
  const int bt   = blockIdx.x * 4 + w;   // 0..32767
  const int t    = bt & (TT - 1);

  // 1) fp[d,u] = x[bt,d]*kernel[d,u] + bias[d,u]; y = latent[bt,:]
  for (int u = lane; u < 64; u += 32) yS[w][u] = (float)latent[bt * 64 + u];
  for (int idx = lane; idx < 48 * 64; idx += 32) {
    const int d = idx >> 6, u = idx & 63;
    float v = 0.f;
    if (d < DFF) v = x[bt * DFF + d] * kern[d * 64 + u] + pbias[d * 64 + u];
    fpH[w][idx] = (_Float16)v;
  }
  __syncthreads();

  // 2) XTy[d] = sum_u fp[d,u]*y[u]
  for (int d = lane; d < 48; d += 32) {
    float s = 0.f;
    for (int u = 0; u < 64; ++u) s += (float)fpH[w][d * 64 + u] * yS[w][u];
    rhs[w][d] = s;
  }

  // 3) G = fp * fp^T via 3x3 WMMA tiles (K=64); B^T == fp itself.
  const int nl = lane & 15;
  const int mb = (lane & 16) ? 8 : 0;
  for (int mt = 0; mt < 3; ++mt) {
    for (int nt = 0; nt < 3; ++nt) {
      v8f acc = {};
#pragma unroll
      for (int ks = 0; ks < 64; ks += 32) {
        v16h a = frag_rowmajor(fpH[w], 64, mt * 16, ks);
        v16h b = frag_rowmajor(fpH[w], 64, nt * 16, ks);
        acc = __builtin_amdgcn_wmma_f32_16x16x32_f16(false, a, false, b,
                                                     (short)0, acc, false, false);
      }
      const int n = nt * 16 + nl;
#pragma unroll
      for (int r = 0; r < 8; ++r) {
        const int m = mt * 16 + r + mb;
        if (m < DFF && n < DFF) Gm[w][m * 41 + n] = acc[r];
      }
    }
  }
  __syncthreads();

  // 4) Cholesky: G = L L^T (in-place lower triangle), one wave per instance.
  float* Gw = Gm[w];
  for (int j = 0; j < DFF; ++j) {
    if (lane == 0) {
      float s = Gw[j * 41 + j];
      for (int k = 0; k < j; ++k) s -= Gw[j * 41 + k] * Gw[j * 41 + k];
      Gw[j * 41 + j] = __builtin_amdgcn_sqrtf(fmaxf(s, 1e-20f));
    }
    __syncthreads();
    const float rinv = fast_rcp(Gw[j * 41 + j]);
    for (int i = j + 1 + lane; i < DFF; i += 32) {
      float s = Gw[i * 41 + j];
      for (int k = 0; k < j; ++k) s -= Gw[i * 41 + k] * Gw[j * 41 + k];
      Gw[i * 41 + j] = s * rinv;
    }
    __syncthreads();
  }

  // 5) Forward solve L v = XTy  (in rhs)
  for (int j = 0; j < DFF; ++j) {
    if (lane == 0) rhs[w][j] *= fast_rcp(Gw[j * 41 + j]);
    __syncthreads();
    const float vj = rhs[w][j];
    for (int i = j + 1 + lane; i < DFF; i += 32) rhs[w][i] -= Gw[i * 41 + j] * vj;
    __syncthreads();
  }
  // 6) Backward solve L^T alpha = v (in rhs)
  for (int j = DFF - 1; j >= 0; --j) {
    if (lane == 0) rhs[w][j] *= fast_rcp(Gw[j * 41 + j]);
    __syncthreads();
    const float aj = rhs[w][j];
    for (int i = lane; i < j; i += 32) rhs[w][i] -= Gw[j * 41 + i] * aj;
    __syncthreads();
  }

  // 7) ohat[u] = beta[t] * sum_d alpha[d] * fp[d,u]
  const float bt_w = betaN[t];
  for (int u = lane; u < 64; u += 32) {
    float s = 0.f;
    for (int d = 0; d < DFF; ++d) s += rhs[w][d] * (float)fpH[w][d * 64 + u];
    ohat[bt * 64 + u] = bt_w * s;
  }
}

// ---- K7: z[b,u] = sum_t ohat[b,t,u] (fixed order -> deterministic) --------
__global__ __launch_bounds__(256) void reduce_kernel(const float* __restrict__ ohat,
                                                     float* __restrict__ z) {
  const int idx = blockIdx.x * 256 + threadIdx.x;  // 0..16383
  const int b = idx >> 6, u = idx & 63;
  float s = 0.f;
  for (int t = 0; t < TT; ++t) s += ohat[(b * TT + t) * 64 + u];
  z[idx] = s;
}

extern "C" void kernel_launch(void* const* d_in, const int* in_sizes, int n_in,
                              void* d_out, int out_size, void* d_ws, size_t ws_size,
                              hipStream_t stream) {
  (void)in_sizes; (void)n_in; (void)out_size; (void)ws_size;
  const float* x    = (const float*)d_in[0];
  const float* W0   = (const float*)d_in[1];
  const float* U0   = (const float*)d_in[2];
  const float* b0   = (const float*)d_in[3];
  const float* W1   = (const float*)d_in[4];
  const float* U1   = (const float*)d_in[5];
  const float* b1   = (const float*)d_in[6];
  const float* kern = (const float*)d_in[7];
  const float* pbias= (const float*)d_in[8];
  const float* bw   = (const float*)d_in[9];
  float* zout = (float*)d_out;

  char* ws = (char*)d_ws;
  float*     bufA  = (float*)ws;                              // 33.5 MB: XW0 -> HW1 -> ohat
  size_t off = (size_t)NBT * G4 * 4;
  _Float16*  bufBh = (_Float16*)(ws + off); off += (size_t)NBT * HH * 2;  // h_seq -> latent
  _Float16*  xpadH = (_Float16*)(ws + off); off += (size_t)NBT * 64 * 2;  // padded f16 x
  _Float16*  W0T   = (_Float16*)(ws + off); off += (size_t)G4 * 64 * 2;
  _Float16*  W1T   = (_Float16*)(ws + off); off += (size_t)G4 * 64 * 2;
  float*     betaN = (float*)(ws + off);

  pack_rows_kernel<<<8192, 256, 0, stream>>>(x, xpadH, DFF);
  packT_kernel<<<64, 256, 0, stream>>>(W0, W0T, DFF);
  packT_kernel<<<64, 256, 0, stream>>>(W1, W1T, HH);
  gemm_bias_kernel<<<256, 256, 0, stream>>>(xpadH, W0T, b0, bufA);   // XW0
  lstm_kernel<<<16, 512, 0, stream>>>(bufA, U0, bufBh);              // h_seq (f16)
  gemm_bias_kernel<<<256, 256, 0, stream>>>(bufBh, W1T, b1, bufA);   // HW1
  lstm_kernel<<<16, 512, 0, stream>>>(bufA, U1, bufBh);              // latent (f16)
  beta_kernel<<<1, 128, 0, stream>>>(bw, betaN);
  attn_kernel<<<8192, 128, 0, stream>>>(x, bufBh, kern, pbias, betaN, bufA);
  reduce_kernel<<<64, 256, 0, stream>>>(bufA, zout);
}
